// SparseAttention_90915867722243
// MI455X (gfx1250) — matrix operand order repaired and checked by reading, not compiled, not run on hardware
//
#include <hip/hip_runtime.h>
#include <hip/hip_bf16.h>

// ---------------------------------------------------------------------------
// Sliding-window attention (B=2, S=2048, HIDDEN=1024, HEADS=16, HEAD_DIM=64,
// WINDOW=256) for gfx1250 using v_wmma_f32_16x16x32_bf16 + TDM staging.
// Outputs: out [B,S,1024] fp32, then weights [B,16,S,S] fp32, concatenated.
// ---------------------------------------------------------------------------

typedef __bf16 bf16_t;
typedef __attribute__((ext_vector_type(16))) __bf16 v16bf;
typedef __attribute__((ext_vector_type(8)))  __bf16 v8bf;
typedef __attribute__((ext_vector_type(8)))  float  v8f;
typedef __attribute__((ext_vector_type(4)))  unsigned int v4u;
typedef __attribute__((ext_vector_type(8)))  int v8i;
typedef __attribute__((ext_vector_type(4)))  int v4i;

#define S_LEN   2048
#define HID     1024
#define NHEADS  16
#define HDIM    64
#define HWIN    128          // attend to |i-j| <= 128
#define BATCH   2
#define GK      1024         // GEMM K  (== HID)
#define GN      1024         // GEMM N  (== HID)
#define GM      4096         // GEMM M  (== BATCH*S_LEN)

#define CW      272          // 17 key tiles * 16
#define CP      288          // CW padded to multiple of 32 (zero-padded P)
#define GC      336          // staged V columns per 64-query group
#define NWAVE   4            // waves per attention block

#define OUT_ELEMS ((size_t)BATCH * S_LEN * HID)

__device__ __forceinline__ int iminD(int a, int b) { return a < b ? a : b; }
__device__ __forceinline__ int imaxD(int a, int b) { return a > b ? a : b; }

// Build a 16x32 bf16 A-fragment: lane<16 holds K[base..base+15];
// lane>=16 holds K[base+16..base+31]. Caller passes the
// per-lane base pointer (already offset by 0 or 16).
__device__ __forceinline__ v16bf load_afrag(const bf16_t* p) {
    v8bf c0 = *(const v8bf*)(p);
    v8bf c1 = *(const v8bf*)(p + 8);
    v16bf a;
#pragma unroll
    for (int i = 0; i < 8; ++i) { a[i] = c0[i]; a[8 + i] = c1[i]; }
    return a;
}

__device__ __forceinline__ v8f wmma_bf16(v16bf a, v16bf b, v8f c) {
    return __builtin_amdgcn_wmma_f32_16x16x32_bf16(
        /*neg_a=*/false, a, /*neg_b=*/false, b,
        /*c_mod=*/(short)0, c, /*reuse_a=*/false, /*reuse_b=*/false);
}

// ---------------------------------------------------------------------------
// Utility kernels
// ---------------------------------------------------------------------------
__global__ void k_zero_f4(float4* __restrict__ p, size_t n4) {
    size_t i = (size_t)blockIdx.x * blockDim.x + threadIdx.x;
    size_t stride = (size_t)gridDim.x * blockDim.x;
    float4 z; z.x = 0.f; z.y = 0.f; z.z = 0.f; z.w = 0.f;
    for (; i < n4; i += stride) p[i] = z;
}

__global__ void k_cvt_bf16(const float* __restrict__ x, bf16_t* __restrict__ y, int n) {
    int i = blockIdx.x * blockDim.x + threadIdx.x;
    int stride = gridDim.x * blockDim.x;
    for (; i < n; i += stride) y[i] = (bf16_t)x[i];
}

// Wt[n*dim + k] = bf16(W[k*dim + n])  (so B-fragments read contiguous K runs)
__global__ void k_transpose_bf16(const float* __restrict__ W, bf16_t* __restrict__ Wt, int dim) {
    int i = blockIdx.x * blockDim.x + threadIdx.x;
    int stride = gridDim.x * blockDim.x;
    int n = dim * dim;
    for (; i < n; i += stride) {
        int k = i / dim, c = i % dim;
        Wt[(size_t)c * dim + k] = (bf16_t)W[i];
    }
}

// ---------------------------------------------------------------------------
// bf16 WMMA GEMM:  C[4096,1024] = scale * (A @ B + bias),  Bt[N][K] given.
// Block = 8 waves = 128 rows x one 64-col group.  The block's whole
// Bt[64][1024] slab (128 KB) is DMA'd to LDS once via TENSOR_LOAD_TO_LDS
// (TDM, TENSORcnt) and B-fragments are then read from LDS; A-fragments are
// read straight from global (L2-resident).
// ---------------------------------------------------------------------------
__global__ __launch_bounds__(256)
void k_gemm_bf16(const bf16_t* __restrict__ A, const bf16_t* __restrict__ Bt,
                 const float* __restrict__ bias,
                 float* __restrict__ outF, bf16_t* __restrict__ outB,
                 float scale) {
    extern __shared__ char smem[];
    bf16_t* Bl = (bf16_t*)smem;                     // [64][GK] = 128 KB

    const int ng   = blockIdx.x & 15;               // 16 column groups
    const int msup = blockIdx.x >> 4;               // 32 row supertiles (128 rows)
    const bf16_t* gsrc = Bt + (size_t)(ng * 64) * GK;   // contiguous 64*GK slab

#if __has_builtin(__builtin_amdgcn_tensor_load_to_lds)
    if (threadIdx.x < 32) {                         // wave 0 issues the DMA
        typedef __bf16 __attribute__((address_space(3)))* lds_p;
        unsigned long long ga = (unsigned long long)(uintptr_t)gsrc;
        unsigned int la = (unsigned int)(uintptr_t)(lds_p)Bl;
        // D# group0: count=1 | lds_addr | global_addr | type=2 ("image")
        v4u g0 = { 1u, la, (unsigned int)ga,
                   (unsigned int)(ga >> 32) | 0x80000000u };
        // D# group1: data_size=2B; tensor_dim0=1024, tensor_dim1=64;
        //            tile_dim0=1024, tile_dim1=64; dim0_stride=1024
        v8i g1 = { (int)0x00010000,                 // wg_mask=0, data_size=1(2B)
                   (int)0x04000000,                 // tensor_dim0[15:0]<<16
                   (int)0x00400000,                 // dim0 hi | tensor_dim1 lo<<16
                   (int)0x04000000,                 // dim1 hi | tile_dim0<<16
                   64,                              // tile_dim1 (tile_dim2=0)
                   1024,                            // dim0_stride[31:0]
                   0, 0 };
        v4i gz = { 0, 0, 0, 0 };
#if __has_include(<hip/amd_detail/amd_gfx1250_TDM.h>)
        v8i gz8 = { 0, 0, 0, 0, 0, 0, 0, 0 };
        __builtin_amdgcn_tensor_load_to_lds(g0, g1, gz, gz, gz8, 0);
#else
        __builtin_amdgcn_tensor_load_to_lds(g0, g1, gz, gz, 0);
#endif
        __builtin_amdgcn_s_wait_tensorcnt(0);
    }
#else
    // Portable fallback: cooperative vector copy (slab is contiguous).
    {
        const v8bf* s = (const v8bf*)gsrc;
        v8bf* d = (v8bf*)Bl;
        for (int i = threadIdx.x; i < 64 * GK / 8; i += 256) d[i] = s[i];
    }
#endif
    __syncthreads();

    const int wv   = threadIdx.x >> 5;
    const int lane = threadIdx.x & 31;
    const int l15  = lane & 15;
    const bool lo  = lane < 16;

    const bf16_t* Arow = A + (size_t)(msup * 128 + wv * 16 + l15) * GK + (lo ? 0 : 16);
    const bf16_t* B0 = Bl + (size_t)( 0 + l15) * GK + (lo ? 0 : 16);
    const bf16_t* B1 = Bl + (size_t)(16 + l15) * GK + (lo ? 0 : 16);
    const bf16_t* B2 = Bl + (size_t)(32 + l15) * GK + (lo ? 0 : 16);
    const bf16_t* B3 = Bl + (size_t)(48 + l15) * GK + (lo ? 0 : 16);

    v8f acc0 = {}, acc1 = {}, acc2 = {}, acc3 = {};
    for (int kk = 0; kk < GK; kk += 32) {
        v16bf a = load_afrag(Arow + kk);
        acc0 = wmma_bf16(a, *(const v16bf*)(B0 + kk), acc0);
        acc1 = wmma_bf16(a, *(const v16bf*)(B1 + kk), acc1);
        acc2 = wmma_bf16(a, *(const v16bf*)(B2 + kk), acc2);
        acc3 = wmma_bf16(a, *(const v16bf*)(B3 + kk), acc3);
    }

    const int rbase = msup * 128 + wv * 16 + (lo ? 0 : 8);
#pragma unroll
    for (int t = 0; t < 4; ++t) {
        v8f acc = (t == 0) ? acc0 : (t == 1) ? acc1 : (t == 2) ? acc2 : acc3;
        const int col = ng * 64 + t * 16 + l15;
        const float bn = bias ? bias[col] : 0.f;
#pragma unroll
        for (int r = 0; r < 8; ++r) {
            float v = (acc[r] + bn) * scale;
            size_t idx = (size_t)(rbase + r) * GN + col;
            if (outF) outF[idx] = v;
            if (outB) outB[idx] = (bf16_t)v;
        }
    }
}

// ---------------------------------------------------------------------------
// Sliding-window attention core. One block = (batch, head, 64-query group),
// 4 waves, each wave owns a 16-query tile and its 272-key window.
// ---------------------------------------------------------------------------
__global__ __launch_bounds__(128)
void k_attn(const bf16_t* __restrict__ Q, const bf16_t* __restrict__ Kb,
            const bf16_t* __restrict__ Vb,
            float* __restrict__ wout, bf16_t* __restrict__ oatt) {
    const int qg  = blockIdx.x % (S_LEN / 64);
    const int bh  = blockIdx.x / (S_LEN / 64);
    const int b   = bh / NHEADS;
    const int h   = bh % NHEADS;
    const int q0g = qg * 64;

    extern __shared__ char sm[];
    bf16_t* Vl = (bf16_t*)sm;                                  // [HDIM][GC]
    float*  Sl = (float*)(sm + (size_t)HDIM * GC * 2);         // [NWAVE][16][CP]
    bf16_t* Pl = (bf16_t*)(sm + (size_t)HDIM * GC * 2
                              + (size_t)NWAVE * 16 * CP * 4);  // [NWAVE][16][CP]

    const int tid = threadIdx.x;

    // Stage V transposed: Vl[d][c] = V[clamp(q0g-128+c)][h*64+d]
    for (int c = tid; c < GC; c += 128) {
        int j = iminD(imaxD(q0g - HWIN + c, 0), S_LEN - 1);
        const bf16_t* vrow = Vb + (size_t)(b * S_LEN + j) * HID + h * HDIM;
#pragma unroll
        for (int dd = 0; dd < 8; ++dd) {
            v8bf v = *(const v8bf*)(vrow + dd * 8);
#pragma unroll
            for (int e = 0; e < 8; ++e) Vl[(dd * 8 + e) * GC + c] = v[e];
        }
    }
    __syncthreads();

    const int w    = tid >> 5;
    const int lane = tid & 31;
    const int l15  = lane & 15;
    const bool loL = lane < 16;
    const int q0    = q0g + w * 16;
    const int jbase = q0 - HWIN;
    const int vb    = w * 16;           // wave's column offset inside Vl
    float*  Sw = Sl + (size_t)w * 16 * CP;
    bf16_t* Pw = Pl + (size_t)w * 16 * CP;

    // ---- scores: Q(16x64) @ K^T (64 x 272), 17 key tiles x 2 K-steps ----
    const bf16_t* qrow = Q + (size_t)(b * S_LEN + q0 + l15) * HID + h * HDIM;
    v16bf a0 = load_afrag(qrow + (loL ? 0 : 16));         // d = 0..31
    v16bf a1 = load_afrag(qrow + 32 + (loL ? 0 : 16));    // d = 32..63

    for (int kt = 0; kt < 17; ++kt) {
        int j  = jbase + kt * 16 + l15;
        int jc = iminD(imaxD(j, 0), S_LEN - 1);
        const bf16_t* krow = Kb + (size_t)(b * S_LEN + jc) * HID + h * HDIM + (loL ? 0 : 16);
        v16bf b0 = *(const v16bf*)(krow);
        v16bf b1 = *(const v16bf*)(krow + 32);
        v8f acc = {};
        acc = wmma_bf16(a0, b0, acc);
        acc = wmma_bf16(a1, b1, acc);
        const int col = kt * 16 + l15;
        const int rb  = loL ? 0 : 8;
#pragma unroll
        for (int r = 0; r < 8; ++r) Sw[(rb + r) * CP + col] = acc[r];
    }
    __syncthreads();

    // ---- softmax (2 lanes per row, 136 columns each) ----
    const int row  = l15;
    const int half = lane >> 4;
    const int i    = q0 + row;
    const int lo_c = imaxD(row, -jbase);                       // j>=0 and j>=i-128
    const int hi_c = iminD(row + 2 * HWIN, (S_LEN - 1) - jbase);
    const int cb = half * 136, ce = cb + 136;

    float m = -__builtin_inff();
    for (int c = cb; c < ce; ++c)
        if (c >= lo_c && c <= hi_c) m = fmaxf(m, Sw[row * CP + c]);
    m = fmaxf(m, __shfl_xor(m, 16, 32));

    float sum = 0.f;
    for (int c = cb; c < ce; ++c) {
        if (c >= lo_c && c <= hi_c) {
            float e = __expf(Sw[row * CP + c] - m);
            Sw[row * CP + c] = e;
            sum += e;
        }
    }
    sum += __shfl_xor(sum, 16, 32);
    const float inv = 1.f / sum;

    float* wrow = wout + ((size_t)((b * NHEADS + h) * S_LEN + i)) * S_LEN;
    for (int c = cb; c < ce; ++c) {
        bool val = (c >= lo_c && c <= hi_c);
        float p = val ? Sw[row * CP + c] * inv : 0.f;
        Pw[row * CP + c] = (bf16_t)p;
        if (val) wrow[jbase + c] = p;       // off-band already zero-filled
    }
    if (half) {                             // zero pad columns CW..CP-1
        for (int c = CW; c < CP; ++c) Pw[row * CP + c] = (bf16_t)0.f;
    }
    __syncthreads();

    // ---- O(16x64) = P(16x288) @ V(288x64) ----
#pragma unroll
    for (int t = 0; t < 4; ++t) {
        v8f acc = {};
        const bf16_t* arow = Pw + (size_t)l15 * CP + (loL ? 0 : 16);
        const bf16_t* bcol = Vl + (size_t)(t * 16 + l15) * GC + vb + (loL ? 0 : 16);
#pragma unroll
        for (int ks = 0; ks < 9; ++ks) {
            int kk = ks * 32;
            v16bf a  = load_afrag(arow + kk);
            v16bf bv = *(const v16bf*)(bcol + kk);
            acc = wmma_bf16(a, bv, acc);
        }
        const int rb = loL ? 0 : 8;
#pragma unroll
        for (int r = 0; r < 8; ++r) {
            int orow = q0 + rb + r;
            oatt[(size_t)(b * S_LEN + orow) * HID + h * HDIM + t * 16 + l15] = (bf16_t)acc[r];
        }
    }
}

// ---------------------------------------------------------------------------
// Host launcher
// ---------------------------------------------------------------------------
extern "C" void kernel_launch(void* const* d_in, const int* in_sizes, int n_in,
                              void* d_out, int out_size, void* d_ws, size_t ws_size,
                              hipStream_t stream) {
    (void)in_sizes; (void)n_in; (void)out_size; (void)ws_size;
    const float* query = (const float*)d_in[0];
    const float* Wq = (const float*)d_in[1]; const float* bq = (const float*)d_in[2];
    const float* Wk = (const float*)d_in[3]; const float* bk = (const float*)d_in[4];
    const float* Wv = (const float*)d_in[5]; const float* bv = (const float*)d_in[6];
    const float* Wo = (const float*)d_in[7]; const float* bo = (const float*)d_in[8];

    const size_t MB = 1u << 20;
    char* ws = (char*)d_ws;
    bf16_t* qbf = (bf16_t*)(ws);              // 8 MB  [4096,1024]
    bf16_t* WqT = (bf16_t*)(ws +  8 * MB);    // 2 MB each (transposed, bf16)
    bf16_t* WkT = (bf16_t*)(ws + 10 * MB);
    bf16_t* WvT = (bf16_t*)(ws + 12 * MB);
    bf16_t* WoT = (bf16_t*)(ws + 14 * MB);
    bf16_t* Qb  = (bf16_t*)(ws + 16 * MB);    // 8 MB each
    bf16_t* Kb  = (bf16_t*)(ws + 24 * MB);
    bf16_t* Vb  = (bf16_t*)(ws + 32 * MB);
    bf16_t* Ab  = (bf16_t*)(ws + 40 * MB);    // attention output, bf16

    float* out  = (float*)d_out;
    float* wout = out + OUT_ELEMS;            // [B,16,S,S] weights region

    const float qscale = 0.125f;              // 1/sqrt(64), folded into Q

    // 1) zero the weights band region (off-band entries must be exactly 0)
    size_t n4 = ((size_t)BATCH * NHEADS * S_LEN * S_LEN) / 4;
    k_zero_f4<<<4096, 256, 0, stream>>>((float4*)wout, n4);

    // 2) bf16 conversions (+ weight transposes so B-fragments are contiguous)
    k_cvt_bf16<<<4096, 256, 0, stream>>>(query, qbf, GM * HID);
    k_transpose_bf16<<<2048, 256, 0, stream>>>(Wq, WqT, HID);
    k_transpose_bf16<<<2048, 256, 0, stream>>>(Wk, WkT, HID);
    k_transpose_bf16<<<2048, 256, 0, stream>>>(Wv, WvT, HID);
    k_transpose_bf16<<<2048, 256, 0, stream>>>(Wo, WoT, HID);

    // 3) projections: 512 blocks x (128 rows x 64 cols); 128 KB LDS B slab
    const size_t gsmem = (size_t)64 * GK * 2;   // 131072 B
    hipFuncSetAttribute((const void*)k_gemm_bf16,
                        hipFuncAttributeMaxDynamicSharedMemorySize, (int)gsmem);
    k_gemm_bf16<<<512, 256, gsmem, stream>>>(qbf, WqT, bq, nullptr, Qb, qscale);
    k_gemm_bf16<<<512, 256, gsmem, stream>>>(qbf, WkT, bk, nullptr, Kb, 1.0f);
    k_gemm_bf16<<<512, 256, gsmem, stream>>>(qbf, WvT, bv, nullptr, Vb, 1.0f);

    // 4) attention: B*H*(S/64) = 1024 blocks, 4 waves, 150 KB dynamic LDS
    const size_t smem = (size_t)HDIM * GC * 2       // V^T     43008 B
                      + (size_t)NWAVE * 16 * CP * 4 // scores 73728 B
                      + (size_t)NWAVE * 16 * CP * 2;// P      36864 B
    hipFuncSetAttribute((const void*)k_attn,
                        hipFuncAttributeMaxDynamicSharedMemorySize, (int)smem);
    k_attn<<<BATCH * NHEADS * (S_LEN / 64), 128, smem, stream>>>(Qb, Kb, Vb, wout, Ab);

    // 5) output projection -> fp32 out
    k_gemm_bf16<<<512, 256, gsmem, stream>>>(Ab, WoT, bo, out, nullptr, 1.0f);
}
